// GATGNN_67448166416735
// MI455X (gfx1250) — matrix operand order repaired and checked by reading, not compiled, not run on hardware
//
#include <hip/hip_runtime.h>
#include <hip/hip_bf16.h>

// ---------------------------------------------------------------------------
// GATGNN forward on MI455X (gfx1250).
// Edge GEMMs ([E,128]@[128,64], the FLOP-dominant op) run on
// v_wmma_f32_16x16x32_f16 (fp16 in, fp32 accumulate). Everything else is
// memory-bound reductions / elementwise work.
// ---------------------------------------------------------------------------

#define NN 50000
#define EE 400000
#define GG 1024
#define NH 64
#define EPS 1e-5f

typedef _Float16 v16h __attribute__((ext_vector_type(16)));
typedef _Float16 h8   __attribute__((ext_vector_type(8)));
typedef float    v8f  __attribute__((ext_vector_type(8)));

__device__ __forceinline__ float softplusf(float x) {
  // numerically stable: max(x,0) + log1p(exp(-|x|))
  return fmaxf(x, 0.f) + log1pf(expf(-fabsf(x)));
}
// order-preserving float<->uint monotone mapping (for atomicMax on floats;
// memset(0) == -inf sentinel)
__device__ __forceinline__ unsigned f2mono(float x) {
  unsigned b = __float_as_uint(x);
  return (b & 0x80000000u) ? ~b : (b | 0x80000000u);
}
__device__ __forceinline__ float mono2f(unsigned m) {
  return __uint_as_float((m & 0x80000000u) ? (m ^ 0x80000000u) : ~m);
}

// --------------------------- embeddings ------------------------------------
__global__ void embed_nodes_kernel(const float* __restrict__ x,
                                   const float* __restrict__ W,
                                   const float* __restrict__ b,
                                   float* __restrict__ h,
                                   _Float16* __restrict__ h16) {
  int t = blockIdx.x * blockDim.x + threadIdx.x;   // N*64 threads
  if (t >= NN * NH) return;
  int n = t >> 6, c = t & 63;
  const float* xr = x + (long)n * 92;
  float s = b[c];
#pragma unroll 4
  for (int k = 0; k < 92; ++k) s += xr[k] * W[k * NH + c];
  h[t] = s;
  h16[t] = (_Float16)s;
}

__global__ void embed_edges_kernel(const float* __restrict__ ea,
                                   const float* __restrict__ W,
                                   const float* __restrict__ b,
                                   _Float16* __restrict__ ea16) {
  int t = blockIdx.x * blockDim.x + threadIdx.x;   // E*64 threads
  if (t >= EE * NH) return;
  int e = t >> 6, c = t & 63;
  const float* er = ea + (long)e * 41;
  float s = b[c];
#pragma unroll
  for (int k = 0; k < 41; ++k) s += er[k] * W[k * NH + c];
  float v = (s > 0.f) ? s : 0.2f * s;              // leaky_relu(0.2)
  ea16[t] = (_Float16)v;
}

// ----------------- edge GEMM + attention logit (WMMA core) -----------------
// One wave handles 16 edges. For pass 0 (hi = [h[dst]|ea]) and pass 1
// (hj = [h[src]|ea]) it computes the 16x64 softplus(GEMM) tile with
// 16x v_wmma_f32_16x16x32_f16, dots it with conv_att via LDS, and for pass 1
// also stores tj to global. alpha_out = softplus(dot_i + dot_j).
__global__ void __launch_bounds__(256)
conv_edge_gemm_kernel(const _Float16* __restrict__ h16,
                      const _Float16* __restrict__ ea16,
                      const int* __restrict__ src,
                      const int* __restrict__ dst,
                      const float* __restrict__ conv_W,   // [3,128,64]
                      const float* __restrict__ conv_att, // [3,1,128]
                      int layer,
                      float* __restrict__ tj_out,         // [E,64]
                      float* __restrict__ alpha_out) {    // [E]
  __shared__ __align__(32) _Float16 sWt[64][128];  // transposed weights [n][k]
  __shared__ float sAtt[128];
  __shared__ float sTile[8][16][64];               // per-wave output tile

  const float* Wl = conv_W + (long)layer * 128 * 64;
  for (int i = threadIdx.x; i < 128 * 64; i += 256) {
    int k = i >> 6, n = i & 63;
    sWt[n][k] = (_Float16)Wl[i];
  }
  for (int i = threadIdx.x; i < 128; i += 256) sAtt[i] = conv_att[layer * 128 + i];
  __syncthreads();

  const int lane = threadIdx.x & 31;
  const int wv   = threadIdx.x >> 5;
  const int m    = lane & 15;       // output/tile row within the 16-edge block
  const int g    = lane >> 4;       // half-wave group
  const int e0   = blockIdx.x * 128 + wv * 16;
  const int em   = e0 + m;

  float alphaAcc = 0.f;

  for (int pass = 0; pass < 2; ++pass) {
    const int* idx = pass ? src : dst;
    const _Float16* rowH = h16 + (long)idx[em] * NH;
    const _Float16* rowE = ea16 + (long)em * NH;

    // A fragments for the 4 K-chunks (lane L holds row L&15;
    // elems i -> K = g*16 + i, contiguous, matching B)
    v16h A[4];
#pragma unroll
    for (int kc = 0; kc < 4; ++kc) {
      const _Float16* rs = (kc < 2) ? rowH : rowE;
      int kb = (kc & 1) * 32;
      A[kc] = *(const v16h*)(rs + kb + g * 16);
    }

#pragma unroll
    for (int ct = 0; ct < 4; ++ct) {
      v8f acc = {};
#pragma unroll
      for (int kc = 0; kc < 4; ++kc) {
        // B layout: lane holds col L&15; elems i -> K = g*16 + i (contig in sWt)
        v16h B = *(const v16h*)(&sWt[ct * 16 + m][kc * 32 + g * 16]);
        acc = __builtin_amdgcn_wmma_f32_16x16x32_f16(
            false, A[kc], false, B, (short)0, acc, false, false);
      }
#pragma unroll
      for (int r = 0; r < 8; ++r) {
        float v = softplusf(acc[r]);          // ti/tj = softplus(GEMM)
        int row = r + g * 8;                  // C layout: VGPR r -> M=r (+8 hi half)
        sTile[wv][row][ct * 16 + m] = v;
        if (pass == 1)
          tj_out[(long)(e0 + row) * NH + ct * 16 + m] = v;
      }
    }
    __syncthreads();
    if (lane < 16) {
      const float* arow = sTile[wv][m];
      const float* ap = sAtt + pass * NH;
      float s = 0.f;
#pragma unroll 8
      for (int c = 0; c < NH; ++c) s += arow[c] * ap[c];
      alphaAcc += s;
    }
    __syncthreads();
  }
  if (lane < 16) alpha_out[em] = softplusf(alphaAcc);
}

// ---------------------- reductions / BN / softmax --------------------------
__global__ void mean_var_kernel(const float* __restrict__ x, int n,
                                float* __restrict__ stats) {  // stats[0]=sum,[1]=sumsq
  float s = 0.f, q = 0.f;
  for (int i = blockIdx.x * blockDim.x + threadIdx.x; i < n;
       i += gridDim.x * blockDim.x) {
    float v = x[i]; s += v; q += v * v;
  }
  __shared__ float ss[256], sq[256];
  ss[threadIdx.x] = s; sq[threadIdx.x] = q;
  __syncthreads();
  for (int o = 128; o > 0; o >>= 1) {
    if (threadIdx.x < o) { ss[threadIdx.x] += ss[threadIdx.x + o];
                           sq[threadIdx.x] += sq[threadIdx.x + o]; }
    __syncthreads();
  }
  if (threadIdx.x == 0) { atomicAdd(stats, ss[0]); atomicAdd(stats + 1, sq[0]); }
}

__global__ void alpha_norm_max_kernel(float* __restrict__ alpha,
                                      const float* __restrict__ stats,
                                      const int* __restrict__ dst,
                                      unsigned* __restrict__ node_max) {
  int e = blockIdx.x * blockDim.x + threadIdx.x;
  if (e >= EE) return;
  float mean = stats[0] * (1.f / EE);
  float var  = stats[1] * (1.f / EE) - mean * mean;
  float a = softplusf((alpha[e] - mean) * rsqrtf(var + EPS));
  alpha[e] = a;
  atomicMax(node_max + dst[e], f2mono(a));
}

__global__ void alpha_exp_sum_kernel(float* __restrict__ alpha,
                                     const unsigned* __restrict__ node_max,
                                     const int* __restrict__ dst,
                                     float* __restrict__ node_sum) {
  int e = blockIdx.x * blockDim.x + threadIdx.x;
  if (e >= EE) return;
  int d = dst[e];
  float ex = expf(alpha[e] - mono2f(node_max[d]));
  alpha[e] = ex;
  atomicAdd(node_sum + d, ex);
}

__global__ void aggregate_kernel(const float* __restrict__ tj,
                                 const float* __restrict__ alpha,
                                 const float* __restrict__ node_sum,
                                 const int* __restrict__ dst,
                                 float* __restrict__ agg) {
  int t = blockIdx.x * blockDim.x + threadIdx.x;   // E*64 threads
  int e = t >> 6, c = t & 63;
  int d = dst[e];
  float w = alpha[e] / node_sum[d];
  atomicAdd(agg + (long)d * NH + c, tj[(long)e * NH + c] * w);
}

__global__ void col_stats_kernel(const float* __restrict__ agg,
                                 const float* __restrict__ bias,
                                 float* __restrict__ colstats) { // [64]=sum,[64..]=sumsq
  int c = blockIdx.x;                              // 64 blocks
  float bc = bias[c];
  float s = 0.f, q = 0.f;
  for (int n = threadIdx.x; n < NN; n += blockDim.x) {
    float v = agg[(long)n * NH + c] + bc; s += v; q += v * v;
  }
  __shared__ float ss[256], sq[256];
  ss[threadIdx.x] = s; sq[threadIdx.x] = q;
  __syncthreads();
  for (int o = 128; o > 0; o >>= 1) {
    if (threadIdx.x < o) { ss[threadIdx.x] += ss[threadIdx.x + o];
                           sq[threadIdx.x] += sq[threadIdx.x + o]; }
    __syncthreads();
  }
  if (threadIdx.x == 0) { colstats[c] = ss[0]; colstats[64 + c] = sq[0]; }
}

__global__ void h_update_kernel(const float* __restrict__ agg,
                                const float* __restrict__ bias,
                                const float* __restrict__ colstats,
                                float* __restrict__ h,
                                _Float16* __restrict__ h16) {
  int t = blockIdx.x * blockDim.x + threadIdx.x;   // N*64 threads
  if (t >= NN * NH) return;
  int c = t & 63;
  float mean = colstats[c] * (1.f / NN);
  float var  = colstats[64 + c] * (1.f / NN) - mean * mean;
  float v = softplusf((agg[t] + bias[c] - mean) * rsqrtf(var + EPS));
  h[t] = v;
  h16[t] = (_Float16)v;
}

// -------------------------- composition head -------------------------------
__global__ void comp_att_kernel(const float* __restrict__ h,
                                const float* __restrict__ gf,
                                const int* __restrict__ batch,
                                const float* __restrict__ Wn,  // [167,32]
                                const float* __restrict__ bn,
                                const float* __restrict__ Wa,  // [32]
                                const float* __restrict__ ba,
                                float* __restrict__ a_node,
                                unsigned* __restrict__ gmax) {
  int n = blockIdx.x * blockDim.x + threadIdx.x;
  if (n >= NN) return;
  int b = batch[n];
  const float* hr = h + (long)n * NH;
  const float* gr = gf + (long)b * 103;
  float acc = ba[0];
#pragma unroll 2
  for (int c = 0; c < 32; ++c) {
    float s = bn[c];
    for (int k = 0; k < NH; ++k)  s += hr[k] * Wn[k * 32 + c];
    for (int k = 0; k < 103; ++k) s += gr[k] * Wn[(NH + k) * 32 + c];
    acc += softplusf(s) * Wa[c];
  }
  a_node[n] = acc;
  atomicMax(gmax + b, f2mono(acc));
}

__global__ void comp_exp_sum_kernel(float* __restrict__ a_node,
                                    const unsigned* __restrict__ gmax,
                                    const int* __restrict__ batch,
                                    float* __restrict__ gsum) {
  int n = blockIdx.x * blockDim.x + threadIdx.x;
  if (n >= NN) return;
  int b = batch[n];
  float ex = expf(a_node[n] - mono2f(gmax[b]));
  a_node[n] = ex;
  atomicAdd(gsum + b, ex);
}

__global__ void pool_kernel(const float* __restrict__ h,
                            const float* __restrict__ a_node,
                            const float* __restrict__ gsum,
                            const int* __restrict__ batch,
                            float* __restrict__ pooled) {
  int t = blockIdx.x * blockDim.x + threadIdx.x;   // N*64 threads
  if (t >= NN * NH) return;
  int n = t >> 6, c = t & 63;
  int b = batch[n];
  float w = a_node[n] / gsum[b];
  atomicAdd(pooled + (long)b * NH + c, h[t] * w);
}

__global__ void fc_out_kernel(const float* __restrict__ pooled,
                              const float* __restrict__ fcW,
                              const float* __restrict__ fcb,
                              float* __restrict__ out) {
  int g = blockIdx.x * blockDim.x + threadIdx.x;
  if (g >= GG) return;
  const float* pr = pooled + (long)g * NH;
  float s = fcb[0];
#pragma unroll 8
  for (int c = 0; c < NH; ++c) s += pr[c] * fcW[c];
  out[g] = s;
}

// ------------------------------- driver ------------------------------------
extern "C" void kernel_launch(void* const* d_in, const int* in_sizes, int n_in,
                              void* d_out, int out_size, void* d_ws, size_t ws_size,
                              hipStream_t stream) {
  const float* x         = (const float*)d_in[0];
  const int*   edge_idx  = (const int*)  d_in[1];
  const float* edge_attr = (const float*)d_in[2];
  const int*   batch     = (const int*)  d_in[3];
  const float* gfeat     = (const float*)d_in[4];
  const float* emb_n_W   = (const float*)d_in[5];
  const float* emb_n_b   = (const float*)d_in[6];
  const float* emb_e_W   = (const float*)d_in[7];
  const float* emb_e_b   = (const float*)d_in[8];
  const float* conv_W    = (const float*)d_in[9];
  const float* conv_att  = (const float*)d_in[10];
  const float* conv_bias = (const float*)d_in[11];
  const float* cn_W      = (const float*)d_in[12];
  const float* cn_b      = (const float*)d_in[13];
  const float* ca_W      = (const float*)d_in[14];
  const float* ca_b      = (const float*)d_in[15];
  const float* fc_W      = (const float*)d_in[16];
  const float* fc_b      = (const float*)d_in[17];
  float* out = (float*)d_out;

  const int* src = edge_idx;        // edge_index[0]
  const int* dst = edge_idx + EE;   // edge_index[1]

  // workspace carve-up (256B aligned slices)
  char* w = (char*)d_ws;
  auto carve = [&](size_t bytes) -> char* {
    char* p = w; w += (bytes + 255) & ~(size_t)255; return p;
  };
  float*    h        = (float*)   carve((size_t)NN * NH * 4);
  _Float16* h16      = (_Float16*)carve((size_t)NN * NH * 2);
  _Float16* ea16     = (_Float16*)carve((size_t)EE * NH * 2);
  float*    tj       = (float*)   carve((size_t)EE * NH * 4);
  float*    alpha    = (float*)   carve((size_t)EE * 4);
  unsigned* node_max = (unsigned*)carve((size_t)NN * 4);
  float*    node_sum = (float*)   carve((size_t)NN * 4);
  float*    agg      = (float*)   carve((size_t)NN * NH * 4);
  float*    stats    = (float*)   carve(2 * 4);
  float*    colstats = (float*)   carve(128 * 4);
  float*    a_node   = (float*)   carve((size_t)NN * 4);
  unsigned* gmax     = (unsigned*)carve((size_t)GG * 4);
  float*    gsum     = (float*)   carve((size_t)GG * 4);
  float*    pooled   = (float*)   carve((size_t)GG * NH * 4);

  const int T = 256;

  // embeddings
  embed_nodes_kernel<<<(NN * NH + T - 1) / T, T, 0, stream>>>(x, emb_n_W, emb_n_b, h, h16);
  embed_edges_kernel<<<(EE * NH + T - 1) / T, T, 0, stream>>>(edge_attr, emb_e_W, emb_e_b, ea16);

  // 3 AGATConv layers
  for (int l = 0; l < 3; ++l) {
    hipMemsetAsync(stats,    0, 2 * 4,                stream);
    hipMemsetAsync(node_max, 0, (size_t)NN * 4,       stream);  // mono(-inf)
    hipMemsetAsync(node_sum, 0, (size_t)NN * 4,       stream);
    hipMemsetAsync(agg,      0, (size_t)NN * NH * 4,  stream);

    conv_edge_gemm_kernel<<<EE / 128, 256, 0, stream>>>(
        h16, ea16, src, dst, conv_W, conv_att, l, tj, alpha);
    mean_var_kernel<<<1024, 256, 0, stream>>>(alpha, EE, stats);
    alpha_norm_max_kernel<<<(EE + T - 1) / T, T, 0, stream>>>(alpha, stats, dst, node_max);
    alpha_exp_sum_kernel<<<(EE + T - 1) / T, T, 0, stream>>>(alpha, node_max, dst, node_sum);
    aggregate_kernel<<<(EE * NH) / T, T, 0, stream>>>(tj, alpha, node_sum, dst, agg);
    col_stats_kernel<<<64, 256, 0, stream>>>(agg, conv_bias + l * NH, colstats);
    h_update_kernel<<<(NN * NH + T - 1) / T, T, 0, stream>>>(
        agg, conv_bias + l * NH, colstats, h, h16);
  }

  // composition attention + pooling + fc
  hipMemsetAsync(gmax,   0, (size_t)GG * 4,      stream);
  hipMemsetAsync(gsum,   0, (size_t)GG * 4,      stream);
  hipMemsetAsync(pooled, 0, (size_t)GG * NH * 4, stream);

  comp_att_kernel<<<(NN + T - 1) / T, T, 0, stream>>>(
      h, gfeat, batch, cn_W, cn_b, ca_W, ca_b, a_node, gmax);
  comp_exp_sum_kernel<<<(NN + T - 1) / T, T, 0, stream>>>(a_node, gmax, batch, gsum);
  pool_kernel<<<(NN * NH + T - 1) / T, T, 0, stream>>>(h, a_node, gsum, batch, pooled);
  fc_out_kernel<<<(GG + T - 1) / T, T, 0, stream>>>(pooled, fc_W, fc_b, out);
}
